// RotaryEmbeddingWeights_14843406975064
// MI455X (gfx1250) — matrix operand order repaired and checked by reading, not compiled, not run on hardware
//
#include <hip/hip_runtime.h>
#include <hip/hip_bf16.h>
#include <math.h>

#define B_   4
#define S_   1024
#define HID_ 1024
#define H_   16
#define D_   64

typedef __attribute__((ext_vector_type(16))) __bf16 v16bf;
typedef __attribute__((ext_vector_type(8)))  float  v8f;

union FragU { v16bf v; uint4 u[2]; };

__device__ __forceinline__ unsigned short f2bf(float f) {
    unsigned int u = __float_as_uint(f);
    u += 0x7FFFu + ((u >> 16) & 1u);          // round-to-nearest-even
    return (unsigned short)(u >> 16);
}

// ---------------------------------------------------------------------------
// Stage f32 -> bf16 (vectorized float4 -> ushort4). All sizes are %4 == 0.
// ---------------------------------------------------------------------------
__global__ void cvt_f32_bf16(const float* __restrict__ src,
                             unsigned short* __restrict__ dst, int n) {
    int i = (blockIdx.x * blockDim.x + threadIdx.x) * 4;
    if (i < n) {
        float4 f = *(const float4*)(src + i);
        ushort4 o;
        o.x = f2bf(f.x); o.y = f2bf(f.y); o.z = f2bf(f.z); o.w = f2bf(f.w);
        *(ushort4*)(dst + i) = o;
    }
}

// ---------------------------------------------------------------------------
// Fused Q/K projection (bf16 WMMA, f32 accum) + RoPE + (q only) 1/sqrt(D).
// One wave computes a 16(M=seq) x 64(N=full head D) tile so the rotate-half
// partner (d +/- 32) sits in the same lane / same accumulator slot.
// Software-pipelined: fragments double-buffered so loads of K-step k+1
// overlap the 4 WMMAs of step k (distinct live ranges -> partial waitcnts).
// grid = (B*S/16, H, 2), block = 32 (one wave).
// ---------------------------------------------------------------------------
__global__ void __launch_bounds__(32)
qk_proj_rope(const unsigned short* __restrict__ hsb,   // [B*S][HID] bf16
             const unsigned short* __restrict__ Wqb,   // [HID][HID] bf16 (row n, col k)
             const unsigned short* __restrict__ Wkb,
             const float* __restrict__ cosp,           // [S][D]
             const float* __restrict__ sinp,           // [S][D]
             unsigned short* __restrict__ qb,          // [B][H][S][D] bf16, pre-scaled
             unsigned short* __restrict__ kb) {
    const int lane = threadIdx.x;
    const int ln   = lane & 15;
    const int hi   = lane >> 4;
    const int m0   = blockIdx.x * 16;          // flat row in [0, B*S)
    const int h    = blockIdx.y;
    const bool isQ = (blockIdx.z == 0);
    const unsigned short* W    = isQ ? Wqb : Wkb;
    unsigned short*       outp = isQ ? qb  : kb;

    const int arow = (m0 + ln) * HID_;

    FragU fa[2];
    FragU fb[2][4];

    auto loadA = [&](FragU& a, int kbase) {
        a.u[0] = *(const uint4*)(hsb + arow + kbase + hi * 8);
        a.u[1] = *(const uint4*)(hsb + arow + kbase + 16 + hi * 8);
    };
    auto loadB = [&](FragU* b4, int kbase) {
        const int wcol = kbase + hi * 8;
#pragma unroll
        for (int j = 0; j < 4; ++j) {
            const unsigned short* wr =
                W + (size_t)(h * D_ + j * 16 + ln) * HID_ + wcol;
            b4[j].u[0] = *(const uint4*)(wr);
            b4[j].u[1] = *(const uint4*)(wr + 16);
        }
    };

    v8f acc[4] = {{}, {}, {}, {}};
    loadA(fa[0], 0);
    loadB(fb[0], 0);
#pragma unroll
    for (int kt = 0; kt < HID_ / 32; ++kt) {
        const int cur = kt & 1;
        const int nxt = cur ^ 1;
        if (kt + 1 < HID_ / 32) {               // prefetch next K-step
            loadA(fa[nxt], (kt + 1) * 32);
            loadB(fb[nxt], (kt + 1) * 32);
        }
#pragma unroll
        for (int j = 0; j < 4; ++j)
            acc[j] = __builtin_amdgcn_wmma_f32_16x16x32_bf16(
                false, fa[cur].v, false, fb[cur][j].v, (short)0, acc[j],
                false, false);
    }

    // RoPE: d = j*16+ln ; partner d+/-32 -> accumulator j+/-2, same lane/slot.
    const float scale = isQ ? 0.125f : 1.0f;   // fold 1/sqrt(64) into q (exact)
#pragma unroll
    for (int v = 0; v < 8; ++v) {
        const int m = m0 + hi * 8 + v;         // C/D layout: row = hi*8 + v
        const int b = m >> 10;                 // / S_
        const int s = m & (S_ - 1);
        const float* cr = cosp + s * D_;
        const float* sr = sinp + s * D_;
        const float c0 = cr[ln],      c1 = cr[16 + ln];
        const float c2 = cr[32 + ln], c3 = cr[48 + ln];
        const float s0 = sr[ln],      s1 = sr[16 + ln];
        const float s2 = sr[32 + ln], s3 = sr[48 + ln];
        const float r0 = (acc[0][v] * c0 - acc[2][v] * s0) * scale;
        const float r1 = (acc[1][v] * c1 - acc[3][v] * s1) * scale;
        const float r2 = (acc[2][v] * c2 + acc[0][v] * s2) * scale;
        const float r3 = (acc[3][v] * c3 + acc[1][v] * s3) * scale;
        unsigned short* orow = outp + (((size_t)(b * H_ + h) * S_ + s) * D_);
        orow[ln]      = f2bf(r0);
        orow[16 + ln] = f2bf(r1);
        orow[32 + ln] = f2bf(r2);
        orow[48 + ln] = f2bf(r3);
    }
}

// ---------------------------------------------------------------------------
// scores = q . k^T (2 WMMAs per 16x16 tile over D=64) + causal mask + softmax.
// Block = 256 threads = 8 waves; one block owns 16 query rows x all 1024 keys
// for one (b,h). Wave w handles key tiles [8w, 8w+8). Branch-free pipelined
// tile loop: above-diagonal tiles do a dummy (L2-hit) load of tile 0 and are
// masked to -inf afterward, keeping the load->WMMA pipeline straight-line.
// Row reductions: shfl_xor in the 16-lane N-group + LDS combine across waves.
// grid = (S/16, H, B).
// ---------------------------------------------------------------------------
__global__ void __launch_bounds__(256)
attn_softmax(const unsigned short* __restrict__ qb,
             const unsigned short* __restrict__ kb,
             float* __restrict__ out) {
    __shared__ float smax[16 * 8];
    __shared__ float ssum[16 * 8];
    const int tid  = threadIdx.x;
    const int lane = tid & 31;
    const int w    = tid >> 5;
    const int ln   = lane & 15;
    const int hi   = lane >> 4;
    const int m0   = blockIdx.x * 16;
    const int ktmax = m0 >> 4;                 // last key-tile touching diagonal
    const size_t bh = (size_t)(blockIdx.z * H_ + blockIdx.y) * S_;

    // Q fragments (D = 0..31 and 32..63)
    const unsigned short* qrow = qb + (bh + m0 + ln) * D_;
    FragU aq0, aq1;
    aq0.u[0] = *(const uint4*)(qrow + hi * 8);
    aq0.u[1] = *(const uint4*)(qrow + 16 + hi * 8);
    aq1.u[0] = *(const uint4*)(qrow + 32 + hi * 8);
    aq1.u[1] = *(const uint4*)(qrow + 48 + hi * 8);

    FragU kb0[2], kb1[2];
    auto loadK = [&](int t, int buf) {
        const int kt  = w * 8 + t;
        const int n0  = (kt <= ktmax) ? kt * 16 : 0;   // dummy tile 0 if masked
        const unsigned short* krow = kb + (bh + n0 + ln) * D_;
        kb0[buf].u[0] = *(const uint4*)(krow + hi * 8);
        kb0[buf].u[1] = *(const uint4*)(krow + 16 + hi * 8);
        kb1[buf].u[0] = *(const uint4*)(krow + 32 + hi * 8);
        kb1[buf].u[1] = *(const uint4*)(krow + 48 + hi * 8);
    };

    v8f acc[8];
    loadK(0, 0);
#pragma unroll
    for (int t = 0; t < 8; ++t) {
        const int cur = t & 1;
        const int nxt = cur ^ 1;
        if (t + 1 < 8) loadK(t + 1, nxt);      // overlap next tile's loads
        v8f c = {};
        c = __builtin_amdgcn_wmma_f32_16x16x32_bf16(false, aq0.v, false,
                kb0[cur].v, (short)0, c, false, false);
        c = __builtin_amdgcn_wmma_f32_16x16x32_bf16(false, aq1.v, false,
                kb1[cur].v, (short)0, c, false, false);
        const int kt  = w * 8 + t;
        const bool tv = (kt <= ktmax);
        const int ng  = kt * 16 + ln;
#pragma unroll
        for (int v = 0; v < 8; ++v) {
            const int mg = m0 + hi * 8 + v;
            acc[t][v] = (tv && ng <= mg) ? c[v] : -__builtin_inff();
        }
    }

    // --- row max: per-lane over tiles, shfl within 16-lane group, LDS across waves
    float pmax[8];
#pragma unroll
    for (int v = 0; v < 8; ++v) {
        float mx = acc[0][v];
#pragma unroll
        for (int t = 1; t < 8; ++t) mx = fmaxf(mx, acc[t][v]);
#pragma unroll
        for (int off = 1; off < 16; off <<= 1)
            mx = fmaxf(mx, __shfl_xor(mx, off, 32));
        pmax[v] = mx;
    }
    if (ln == 0) {
#pragma unroll
        for (int v = 0; v < 8; ++v) smax[(hi * 8 + v) * 8 + w] = pmax[v];
    }
    __syncthreads();
    float gmax[8];
#pragma unroll
    for (int v = 0; v < 8; ++v) {
        float mx = -__builtin_inff();
#pragma unroll
        for (int ww = 0; ww < 8; ++ww) mx = fmaxf(mx, smax[(hi * 8 + v) * 8 + ww]);
        gmax[v] = mx;
    }

    // --- exp + row sum
#pragma unroll
    for (int v = 0; v < 8; ++v) {
        float sm = 0.f;
#pragma unroll
        for (int t = 0; t < 8; ++t) {
            float e = __expf(acc[t][v] - gmax[v]);   // exp(-inf) == 0 for masked
            acc[t][v] = e;
            sm += e;
        }
#pragma unroll
        for (int off = 1; off < 16; off <<= 1)
            sm += __shfl_xor(sm, off, 32);
        if (ln == 0) ssum[(hi * 8 + v) * 8 + w] = sm;
    }
    __syncthreads();
    float ginv[8];
#pragma unroll
    for (int v = 0; v < 8; ++v) {
        float sm = 0.f;
#pragma unroll
        for (int ww = 0; ww < 8; ++ww) sm += ssum[(hi * 8 + v) * 8 + ww];
        ginv[v] = 1.0f / sm;
    }

    // --- normalized write: lanes 0..15 cover contiguous n -> coalesced 64B runs
#pragma unroll
    for (int t = 0; t < 8; ++t) {
        const int ng = (w * 8 + t) * 16 + ln;
#pragma unroll
        for (int v = 0; v < 8; ++v) {
            const int mg = m0 + hi * 8 + v;
            out[(bh + mg) * S_ + ng] = acc[t][v] * ginv[v];
        }
    }
}

// ---------------------------------------------------------------------------
extern "C" void kernel_launch(void* const* d_in, const int* in_sizes, int n_in,
                              void* d_out, int out_size, void* d_ws, size_t ws_size,
                              hipStream_t stream) {
    const float* hs   = (const float*)d_in[0];
    const float* cosp = (const float*)d_in[1];
    const float* sinp = (const float*)d_in[2];
    // d_in[3] attention_mask: pure causal, computed analytically in-kernel
    const float* Wq   = (const float*)d_in[4];
    const float* Wk   = (const float*)d_in[5];
    float* out = (float*)d_out;

    unsigned short* ws  = (unsigned short*)d_ws;        // 28 MB total bf16 staging
    unsigned short* hsb = ws;                            // B*S*HID
    unsigned short* wqb = hsb + (size_t)B_ * S_ * HID_;  // HID*HID
    unsigned short* wkb = wqb + (size_t)HID_ * HID_;     // HID*HID
    unsigned short* qbf = wkb + (size_t)HID_ * HID_;     // B*H*S*D
    unsigned short* kbf = qbf + (size_t)B_ * H_ * S_ * D_;

    cvt_f32_bf16<<<(B_ * S_ * HID_ / 4 + 255) / 256, 256, 0, stream>>>(hs, hsb, B_ * S_ * HID_);
    cvt_f32_bf16<<<(HID_ * HID_ / 4 + 255) / 256, 256, 0, stream>>>(Wq, wqb, HID_ * HID_);
    cvt_f32_bf16<<<(HID_ * HID_ / 4 + 255) / 256, 256, 0, stream>>>(Wk, wkb, HID_ * HID_);

    qk_proj_rope<<<dim3(B_ * S_ / 16, H_, 2), 32, 0, stream>>>(
        hsb, wqb, wkb, cosp, sinp, qbf, kbf);

    attn_softmax<<<dim3(S_ / 16, H_, B_), 256, 0, stream>>>(qbf, kbf, out);
}
